// SpaceTimePolarSSM_62079457296821
// MI455X (gfx1250) — matrix operand order repaired and checked by reading, not compiled, not run on hardware
//
#include <hip/hip_runtime.h>
#include <math.h>

typedef __attribute__((ext_vector_type(16))) __bf16 v16bf;
typedef __attribute__((ext_vector_type(8)))  float  v8f;
typedef unsigned short ushort_t;
typedef unsigned int   uint_t;

#define WIDTH_  346
#define HEIGHT_ 260

// ---------------- WMMA helpers (bf16, 16x16x32, f32 accum) ----------------
// A operand: 16x32 bf16, rows over lanes 0..15, lanes 0-15 K=0..15, lanes 16-31 K=16..31.
__device__ __forceinline__ v16bf lda_bf16(const __bf16* base, int row0, int ld, int k0, int lane) {
  const int m = lane & 15, half = lane >> 4;
  const __bf16* p = base + (row0 + m) * ld + k0 + 16 * half;
  union { uint_t u[8]; v16bf v; } r;
#pragma unroll
  for (int i = 0; i < 8; ++i) {
    r.u[i]     = *(const uint_t*)(p + 2 * i);
  }
  return r.v;
}
// B operand: 32x16 bf16, cols over lanes 0..15, lanes 0-15 K=0..15, lanes 16-31 K=16..31.
// Bt is the weight stored transposed [N][K] in LDS (i.e. column-major B).
__device__ __forceinline__ v16bf ldb_bf16(const __bf16* base, int n0, int ld, int k0, int lane) {
  const int n = lane & 15, half = lane >> 4;
  const __bf16* p = base + (n0 + n) * ld + k0 + 16 * half;
  union { uint_t u[8]; v16bf v; } r;
#pragma unroll
  for (int i = 0; i < 8; ++i) r.u[i] = *(const uint_t*)(p + 2 * i);
  return r.v;
}
__device__ __forceinline__ v8f wmma_acc_k(const __bf16* A, int lda, int row0,
                                          const __bf16* Bt, int ldb, int col0,
                                          int kchunks, int lane) {
  v8f acc = {0.f, 0.f, 0.f, 0.f, 0.f, 0.f, 0.f, 0.f};
  for (int kk = 0; kk < kchunks; ++kk) {
    v16bf a = lda_bf16(A, row0, lda, kk * 32, lane);
    v16bf b = ldb_bf16(Bt, col0, ldb, kk * 32, lane);
    acc = __builtin_amdgcn_wmma_f32_16x16x32_bf16(false, a, false, b, (short)0, acc, false, false);
  }
  return acc;
}

// ---------------- Async global->LDS (gfx1250 VGLOBAL async encoding) ----------------
// global_load_async_to_lds_b64: per lane, copies 8 bytes from the global address to the LDS address.
// Tracked by ASYNCcnt. VDST = LDS byte address VGPR; VADDR = 64-bit global address; no SADDR.
__device__ __forceinline__ void async_g2l_b64(unsigned lds_byte_addr, const void* gaddr) {
  asm volatile("global_load_async_to_lds_b64 %0, %1, off"
               :: "v"(lds_byte_addr), "v"(gaddr)
               : "memory");
}
__device__ __forceinline__ void wait_async0_() {
#if __has_builtin(__builtin_amdgcn_s_wait_asynccnt)
  __builtin_amdgcn_s_wait_asynccnt(0);
#else
  asm volatile("s_wait_asynccnt 0x0" ::: "memory");
#endif
}

__device__ __forceinline__ float sigmoidf_(float x) { return 1.f / (1.f + expf(-x)); }
__device__ __forceinline__ float siluf_(float x)    { return x * sigmoidf_(x); }
__device__ __forceinline__ float softplusf_(float x){ return (x > 20.f) ? x : log1pf(expf(x)); }
__device__ __forceinline__ float geluf_(float x)    { return 0.5f * x * (1.f + erff(x * 0.70710678118654752f)); }
__device__ __forceinline__ ushort_t bf16bits_(float x) { __bf16 h = (__bf16)x; return __builtin_bit_cast(ushort_t, h); }

struct MambaP {
  const float *A_log, *D, *conv_b, *conv_w, *dt_bias, *in_proj, *norm_w, *out_proj;
};

// ---------------- Adapter: mag/phase features -> gelu -> LN, bf16 out ----------------
// 32 tokens per block, 256 threads (8 waves). K padded 66->96, 130->160.
#define AD_SMEM (32*96*2 + 32*160*2 + 128*96*2 + 128*160*2 + 32*128*4 + 32*128*4)
__global__ __launch_bounds__(256, 1) void k_adapter(
    const float* __restrict__ zr, const float* __restrict__ zi, const int* __restrict__ coords,
    const float* mag_w, const float* mag_b, const float* nm_g, const float* nm_b,
    const float* ph_w,  const float* ph_b,  const float* np_g, const float* np_b,
    ushort_t* __restrict__ f_mag, ushort_t* __restrict__ f_phase)
{
  extern __shared__ char smem[];
  __bf16* Am = (__bf16*)smem;          // [32][96]
  __bf16* Ap = Am + 32 * 96;           // [32][160]
  __bf16* Wm = Ap + 32 * 160;          // [128][96]  (mag_w^T, zero-padded K)
  __bf16* Wp = Wm + 128 * 96;          // [128][160] (ph_w^T)
  float*  Om = (float*)(Wp + 128 * 160); // [32][128]
  float*  Op = Om + 32 * 128;            // [32][128]

  const int tid = threadIdx.x, lane = tid & 31, wv = tid >> 5;
  __builtin_prefetch(mag_w, 0, 1);
  __builtin_prefetch(ph_w, 0, 1);

  for (int e = tid; e < 128 * 96; e += 256) {
    int n = e / 96, k = e % 96;
    Wm[e] = (k < 66) ? (__bf16)mag_w[k * 128 + n] : (__bf16)0.f;
  }
  for (int e = tid; e < 128 * 160; e += 256) {
    int n = e / 160, k = e % 160;
    Wp[e] = (k < 130) ? (__bf16)ph_w[k * 128 + n] : (__bf16)0.f;
  }
  if (tid < 32) {
    size_t g = (size_t)blockIdx.x * 32 + tid;
    const float* zrp = zr + g * 64;
    const float* zip = zi + g * 64;
    float r[64], ss = 0.f;
#pragma unroll 8
    for (int i = 0; i < 64; ++i) {
      float a = zrp[i], b = zip[i];
      float rv = sqrtf(a * a + b * b);
      r[i] = rv; ss += rv * rv;
    }
    float nrm = fmaxf(sqrtf(ss), 1e-12f);
    int cx = coords[g * 2 + 0], cy = coords[g * 2 + 1];
    float xn = (float)cx / (float)WIDTH_, yn = (float)cy / (float)HEIGHT_;
#pragma unroll 8
    for (int i = 0; i < 64; ++i) Am[tid * 96 + i] = (__bf16)(r[i] / nrm);
    Am[tid * 96 + 64] = (__bf16)xn; Am[tid * 96 + 65] = (__bf16)yn;
    for (int i = 66; i < 96; ++i) Am[tid * 96 + i] = (__bf16)0.f;
#pragma unroll 8
    for (int i = 0; i < 64; ++i) {
      float rs = fmaxf(r[i], 1e-12f);
      Ap[tid * 160 + i]      = (__bf16)(zrp[i] / rs);
      Ap[tid * 160 + 64 + i] = (__bf16)(zip[i] / rs);
    }
    Ap[tid * 160 + 128] = (__bf16)xn; Ap[tid * 160 + 129] = (__bf16)yn;
    for (int i = 130; i < 160; ++i) Ap[tid * 160 + i] = (__bf16)0.f;
  }
  __syncthreads();

  const int n_ = lane & 15, half_ = lane >> 4;
  for (int rt = 0; rt < 2; ++rt) {                 // mag GEMM [32x96]@[96x128]
    v8f acc = wmma_acc_k(Am, 96, rt * 16, Wm, 96, wv * 16, 3, lane);
#pragma unroll
    for (int r = 0; r < 8; ++r)
      Om[(rt * 16 + r + 8 * half_) * 128 + wv * 16 + n_] = acc[r];
  }
  for (int rt = 0; rt < 2; ++rt) {                 // phase GEMM [32x160]@[160x128]
    v8f acc = wmma_acc_k(Ap, 160, rt * 16, Wp, 160, wv * 16, 5, lane);
#pragma unroll
    for (int r = 0; r < 8; ++r)
      Op[(rt * 16 + r + 8 * half_) * 128 + wv * 16 + n_] = acc[r];
  }
  __syncthreads();

  if (tid < 32) {
    size_t g = (size_t)blockIdx.x * 32 + tid;
    // mag: gelu + LN
    float m = 0.f;
    for (int c = 0; c < 128; ++c) { float v = geluf_(Om[tid * 128 + c] + mag_b[c]); Om[tid * 128 + c] = v; m += v; }
    m *= (1.f / 128.f);
    float var = 0.f;
    for (int c = 0; c < 128; ++c) { float d = Om[tid * 128 + c] - m; var += d * d; }
    float inv = rsqrtf(var * (1.f / 128.f) + 1e-5f);
    for (int c = 0; c < 128; ++c)
      f_mag[g * 128 + c] = bf16bits_((Om[tid * 128 + c] - m) * inv * nm_g[c] + nm_b[c]);
    // phase
    m = 0.f;
    for (int c = 0; c < 128; ++c) { float v = geluf_(Op[tid * 128 + c] + ph_b[c]); Op[tid * 128 + c] = v; m += v; }
    m *= (1.f / 128.f);
    var = 0.f;
    for (int c = 0; c < 128; ++c) { float d = Op[tid * 128 + c] - m; var += d * d; }
    inv = rsqrtf(var * (1.f / 128.f) + 1e-5f);
    for (int c = 0; c < 128; ++c)
      f_phase[g * 128 + c] = bf16bits_((Op[tid * 128 + c] - m) * inv * np_g[c] + np_b[c]);
  }
}

// ---------------- Per-segment stable argsort (bitonic, tie-break by index) ----------------
__global__ __launch_bounds__(256, 1) void k_sort(const int* __restrict__ coords,
                                                 uint_t* __restrict__ idx0, uint_t* __restrict__ idx1)
{
  __shared__ uint_t a[256];
  const int s = blockIdx.x, t = threadIdx.x;
  int cx = coords[((size_t)s * 256 + t) * 2 + 0];
  int cy = coords[((size_t)s * 256 + t) * 2 + 1];
  uint_t k0 = (uint_t)(cy * WIDTH_ + cx);
  uint_t k1 = (uint_t)(cx * HEIGHT_ + cy);
  for (int pass = 0; pass < 2; ++pass) {
    uint_t key = pass ? k1 : k0;
    a[t] = (key << 8) | (uint_t)t;     // key < 2^17, idx in low 8 bits => stable sort
    __syncthreads();
    for (int k = 2; k <= 256; k <<= 1)
      for (int j = k >> 1; j > 0; j >>= 1) {
        int p = t ^ j;
        if (p > t) {
          uint_t v0 = a[t], v1 = a[p];
          bool up = ((t & k) == 0);
          if ((v0 > v1) == up) { a[t] = v1; a[p] = v0; }
        }
        __syncthreads();
      }
    uint_t* dst = pass ? idx1 : idx0;
    dst[(size_t)s * 256 + t] = a[t] & 0xFFu;
    __syncthreads();
  }
}

__global__ void k_zero(float* __restrict__ p, size_t n) {
  size_t i = (size_t)blockIdx.x * blockDim.x + threadIdx.x;
  size_t st = (size_t)gridDim.x * blockDim.x;
  for (; i < n; i += st) p[i] = 0.f;
}

// ---------------- Fused local Mamba2 (d_state=16, L=256), one segment per block ----------------
// LDS: xs[256x128]bf16 + xbc[256x304]bf16 + wt[304x128]bf16 + dtv/dAv[256x4]f32 + idx[256]
#define LM_SMEM (256*128*2 + 256*304*2 + 304*128*2 + 256*4*4 + 256*4*4 + 256*4)
__global__ __launch_bounds__(256, 1) void k_local_mamba(
    const ushort_t* __restrict__ xg, const uint_t* __restrict__ idx0, const uint_t* __restrict__ idx1,
    MambaP mp, float* __restrict__ accum)
{
  extern __shared__ char smem[];
  __bf16* xs   = (__bf16*)smem;          // [256][128]  input rows (sorted order), bf16
  __bf16* xbc  = xs + 256 * 128;         // [256][304]  zxbcdt cols 256..547 (+pad)
  __bf16* wt   = xbc + 256 * 304;        // up to [304][128] transposed weights
  float*  dtv  = (float*)(wt + 304 * 128); // [256][4]
  float*  dAv  = dtv + 256 * 4;            // [256][4]
  uint_t* idxL = (uint_t*)(dAv + 256 * 4); // [256]

  const int tid = threadIdx.x, lane = tid & 31, wv = tid >> 5;
  const int seg = blockIdx.x;
  const uint_t* idx = blockIdx.y ? idx1 : idx0;
  const int n_ = lane & 15, half_ = lane >> 4;

  idxL[tid] = idx[(size_t)seg * 256 + tid];
  __builtin_prefetch(mp.in_proj, 0, 1);
  __builtin_prefetch(mp.out_proj, 0, 1);
  __syncthreads();

  // Async DMA gather of sorted rows: one row = 256B = 32 lanes x b64.
  // Fire-and-forget; overlaps with the weight-transpose staging below.
  for (int j = wv; j < 256; j += 8) {
    const ushort_t* src = xg + ((size_t)seg * 256 + idxL[j]) * 128 + lane * 4;
    unsigned ldst = (unsigned)(size_t)(xs + j * 128 + lane * 4);
    async_g2l_b64(ldst, (const void*)src);
  }
  for (int e = tid; e < 304 * 128; e += 256) {       // wt = in_proj[:,256:548]^T (pad->0)
    int n = e >> 7, k = e & 127;
    wt[e] = (n < 292) ? (__bf16)mp.in_proj[(size_t)k * 548 + 256 + n] : (__bf16)0.f;
  }
  wait_async0_();
  __syncthreads();

  // GEMM1: [256x128] @ [128x304] -> xBC (bf16 in LDS) + dt (f32)
  for (int tile = wv; tile < 16 * 19; tile += 8) {
    int r0 = (tile / 19) * 16, c0 = (tile % 19) * 16;
    v8f acc = wmma_acc_k(xs, 128, r0, wt, 128, c0, 4, lane);
#pragma unroll
    for (int r = 0; r < 8; ++r) {
      int row = r0 + r + 8 * half_, col = c0 + n_;
      float v = acc[r];
      if (col < 288)      xbc[row * 304 + col] = (__bf16)v;
      else if (col < 292) dtv[row * 4 + (col - 288)] = v;
    }
  }
  __syncthreads();

  // Depthwise causal conv (width 4) + SiLU, in place, t descending.
  for (int c = tid; c < 288; c += 256) {
    float w0 = mp.conv_w[c * 4 + 0], w1 = mp.conv_w[c * 4 + 1];
    float w2 = mp.conv_w[c * 4 + 2], w3 = mp.conv_w[c * 4 + 3];
    float b = mp.conv_b[c];
    for (int t = 255; t >= 0; --t) {
      float a = b + w3 * (float)xbc[t * 304 + c];
      if (t >= 1) a += w2 * (float)xbc[(t - 1) * 304 + c];
      if (t >= 2) a += w1 * (float)xbc[(t - 2) * 304 + c];
      if (t >= 3) a += w0 * (float)xbc[(t - 3) * 304 + c];
      xbc[t * 304 + c] = (__bf16)siluf_(a);
    }
  }
  // dt = softplus(dt + bias); dA = exp(-exp(A_log)*dt)
  {
#pragma unroll
    for (int h = 0; h < 4; ++h) {
      float sp = softplusf_(dtv[tid * 4 + h] + mp.dt_bias[h]);
      dtv[tid * 4 + h] = sp;
      dAv[tid * 4 + h] = expf(-expf(mp.A_log[h]) * sp);
    }
  }
  __syncthreads();

  // SSM scan: thread = (h,p); 16 states in registers; y overwrites xh column in place.
  {
    const int h = tid >> 6, p = tid & 63;
    const float Dh = mp.D[h];
    float st[16];
#pragma unroll
    for (int n = 0; n < 16; ++n) st[n] = 0.f;
    for (int t = 0; t < 256; ++t) {
      float dtt = dtv[t * 4 + h], dat = dAv[t * 4 + h];
      float x = (float)xbc[t * 304 + h * 64 + p];
      float dtx = dtt * x;
      float y = 0.f;
#pragma unroll
      for (int n = 0; n < 16; ++n) {
        float Bn = (float)xbc[t * 304 + 256 + n];
        float Cn = (float)xbc[t * 304 + 272 + n];
        st[n] = st[n] * dat + dtx * Bn;
        y += st[n] * Cn;
      }
      y += Dh * x;
      xbc[t * 304 + h * 64 + p] = (__bf16)y;   // each thread rewrites only its own element
    }
  }
  __syncthreads();

  // z gate: z = xs @ in_proj[:, :256]; y *= silu(z), in place.
  for (int e = tid; e < 256 * 128; e += 256) {
    int n = e >> 7, k = e & 127;
    wt[e] = (__bf16)mp.in_proj[(size_t)k * 548 + n];
  }
  __syncthreads();
  for (int tile = wv; tile < 16 * 16; tile += 8) {
    int r0 = (tile / 16) * 16, c0 = (tile % 16) * 16;
    v8f acc = wmma_acc_k(xs, 128, r0, wt, 128, c0, 4, lane);
#pragma unroll
    for (int r = 0; r < 8; ++r) {
      int row = r0 + r + 8 * half_, col = c0 + n_;
      float y = (float)xbc[row * 304 + col] * siluf_(acc[r]);
      xbc[row * 304 + col] = (__bf16)y;
    }
  }
  __syncthreads();

  // RMS norm over d_inner=256, * norm_w
  {
    float ss = 0.f;
    for (int c = 0; c < 256; ++c) { float v = (float)xbc[tid * 304 + c]; ss += v * v; }
    float sc = rsqrtf(ss * (1.f / 256.f) + 1e-5f);
    for (int c = 0; c < 256; ++c)
      xbc[tid * 304 + c] = (__bf16)((float)xbc[tid * 304 + c] * sc * mp.norm_w[c]);
  }
  __syncthreads();

  // out_proj GEMM [256x256]@[256x128], scatter back to original token order (accumulate orders)
  for (int e = tid; e < 128 * 256; e += 256) {
    int n = e >> 8, k = e & 255;
    wt[e] = (__bf16)mp.out_proj[(size_t)k * 128 + n];
  }
  __syncthreads();
  for (int tile = wv; tile < 16 * 8; tile += 8) {
    int r0 = (tile / 8) * 16, c0 = (tile % 8) * 16;
    v8f acc = wmma_acc_k(xbc, 304, r0, wt, 256, c0, 8, lane);
#pragma unroll
    for (int r = 0; r < 8; ++r) {
      int row = r0 + r + 8 * half_, col = c0 + n_;
      atomicAdd(&accum[((size_t)seg * 256 + idxL[row]) * 128 + col], acc[r]);
    }
  }
}

// ---------------- segment reduce: (mean+max)/2 of (o0+o1)/2, then LayerNorm ----------------
__global__ __launch_bounds__(128, 1) void k_seg_reduce(const float* __restrict__ accum,
    const float* ln_g, const float* ln_b, float* __restrict__ seg)
{
  __shared__ float sv[128];
  __shared__ float stats[2];
  const int s = blockIdx.x, d = threadIdx.x;
  float m = 0.f, mx = -1e30f;
  for (int j = 0; j < 256; ++j) {
    float v = accum[((size_t)s * 256 + j) * 128 + d] * 0.5f;
    m += v; mx = fmaxf(mx, v);
  }
  float val = (m * (1.f / 256.f) + mx) * 0.5f;
  sv[d] = val;
  __syncthreads();
  if (d == 0) {
    float mean = 0.f;
    for (int i = 0; i < 128; ++i) mean += sv[i];
    mean *= (1.f / 128.f);
    float var = 0.f;
    for (int i = 0; i < 128; ++i) { float t = sv[i] - mean; var += t * t; }
    stats[0] = mean; stats[1] = rsqrtf(var * (1.f / 128.f) + 1e-5f);
  }
  __syncthreads();
  seg[(size_t)s * 128 + d] = (val - stats[0]) * stats[1] * ln_g[d] + ln_b[d];
}

// ---------------- Fused global Mamba2 (d_state=64, L=128), one (batch,stream) per block ----
#define GM_SMEM (128*128*2 + 128*400*2 + 400*128*2 + 128*4*4 + 128*4*4)
__global__ __launch_bounds__(256, 1) void k_global_mamba(
    const float* __restrict__ seg_mag, const float* __restrict__ seg_phase,
    MambaP pm, MambaP pp, float* __restrict__ g_mag, float* __restrict__ g_phase)
{
  extern __shared__ char smem[];
  __bf16* xs  = (__bf16*)smem;           // [128][128]
  __bf16* xbc = xs + 128 * 128;          // [128][400]
  __bf16* wt  = xbc + 128 * 400;         // up to [400][128]
  float*  dtv = (float*)(wt + 400 * 128);  // [128][4]
  float*  dAv = dtv + 128 * 4;             // [128][4]

  const int tid = threadIdx.x, lane = tid & 31, wv = tid >> 5;
  const int b = blockIdx.x;
  const int strm = blockIdx.y;
  const float* sg = strm ? seg_phase : seg_mag;
  MambaP mp = strm ? pp : pm;
  float* gout = strm ? g_phase : g_mag;
  const int n_ = lane & 15, half_ = lane >> 4;

  __builtin_prefetch(mp.in_proj, 0, 1);
  for (int e = tid; e < 128 * 128; e += 256) {
    int t = e >> 7, d = e & 127;
    xs[e] = (__bf16)sg[((size_t)b * 128 + t) * 128 + d];
  }
  for (int e = tid; e < 400 * 128; e += 256) {   // in_proj[:,256:644]^T, pad->0
    int n = e >> 7, k = e & 127;
    wt[e] = (n < 388) ? (__bf16)mp.in_proj[(size_t)k * 644 + 256 + n] : (__bf16)0.f;
  }
  __syncthreads();

  for (int tile = wv; tile < 8 * 25; tile += 8) {      // GEMM1 [128x128]@[128x400]
    int r0 = (tile / 25) * 16, c0 = (tile % 25) * 16;
    v8f acc = wmma_acc_k(xs, 128, r0, wt, 128, c0, 4, lane);
#pragma unroll
    for (int r = 0; r < 8; ++r) {
      int row = r0 + r + 8 * half_, col = c0 + n_;
      float v = acc[r];
      if (col < 384)      xbc[row * 400 + col] = (__bf16)v;
      else if (col < 388) dtv[row * 4 + (col - 384)] = v;
    }
  }
  __syncthreads();

  for (int c = tid; c < 384; c += 256) {               // conv + silu
    float w0 = mp.conv_w[c * 4 + 0], w1 = mp.conv_w[c * 4 + 1];
    float w2 = mp.conv_w[c * 4 + 2], w3 = mp.conv_w[c * 4 + 3];
    float bb = mp.conv_b[c];
    for (int t = 127; t >= 0; --t) {
      float a = bb + w3 * (float)xbc[t * 400 + c];
      if (t >= 1) a += w2 * (float)xbc[(t - 1) * 400 + c];
      if (t >= 2) a += w1 * (float)xbc[(t - 2) * 400 + c];
      if (t >= 3) a += w0 * (float)xbc[(t - 3) * 400 + c];
      xbc[t * 400 + c] = (__bf16)siluf_(a);
    }
  }
  if (tid < 128) {
#pragma unroll
    for (int h = 0; h < 4; ++h) {
      float sp = softplusf_(dtv[tid * 4 + h] + mp.dt_bias[h]);
      dtv[tid * 4 + h] = sp;
      dAv[tid * 4 + h] = expf(-expf(mp.A_log[h]) * sp);
    }
  }
  __syncthreads();

  {                                                    // scan, 64 states / thread
    const int h = tid >> 6, p = tid & 63;
    const float Dh = mp.D[h];
    float st[64];
#pragma unroll
    for (int n = 0; n < 64; ++n) st[n] = 0.f;
    for (int t = 0; t < 128; ++t) {
      float dtt = dtv[t * 4 + h], dat = dAv[t * 4 + h];
      float x = (float)xbc[t * 400 + h * 64 + p];
      float dtx = dtt * x;
      float y = 0.f;
#pragma unroll
      for (int n = 0; n < 64; ++n) {
        float Bn = (float)xbc[t * 400 + 256 + n];
        float Cn = (float)xbc[t * 400 + 320 + n];
        st[n] = st[n] * dat + dtx * Bn;
        y += st[n] * Cn;
      }
      y += Dh * x;
      xbc[t * 400 + h * 64 + p] = (__bf16)y;
    }
  }
  __syncthreads();

  for (int e = tid; e < 256 * 128; e += 256) {         // wt = z-proj^T
    int n = e >> 7, k = e & 127;
    wt[e] = (__bf16)mp.in_proj[(size_t)k * 644 + n];
  }
  __syncthreads();
  for (int tile = wv; tile < 8 * 16; tile += 8) {      // z gate
    int r0 = (tile / 16) * 16, c0 = (tile % 16) * 16;
    v8f acc = wmma_acc_k(xs, 128, r0, wt, 128, c0, 4, lane);
#pragma unroll
    for (int r = 0; r < 8; ++r) {
      int row = r0 + r + 8 * half_, col = c0 + n_;
      float y = (float)xbc[row * 400 + col] * siluf_(acc[r]);
      xbc[row * 400 + col] = (__bf16)y;
    }
  }
  __syncthreads();

  if (tid < 128) {                                     // RMS norm
    float ss = 0.f;
    for (int c = 0; c < 256; ++c) { float v = (float)xbc[tid * 400 + c]; ss += v * v; }
    float sc = rsqrtf(ss * (1.f / 256.f) + 1e-5f);
    for (int c = 0; c < 256; ++c)
      xbc[tid * 400 + c] = (__bf16)((float)xbc[tid * 400 + c] * sc * mp.norm_w[c]);
  }
  __syncthreads();

  for (int e = tid; e < 128 * 256; e += 256) {         // wt = out_proj^T
    int n = e >> 8, k = e & 255;
    wt[e] = (__bf16)mp.out_proj[(size_t)k * 128 + n];
  }
  __syncthreads();
  for (int tile = wv; tile < 8 * 8; tile += 8) {       // out GEMM
    int r0 = (tile / 8) * 16, c0 = (tile % 8) * 16;
    v8f acc = wmma_acc_k(xbc, 400, r0, wt, 256, c0, 8, lane);
#pragma unroll
    for (int r = 0; r < 8; ++r) {
      int row = r0 + r + 8 * half_, col = c0 + n_;
      gout[((size_t)b * 128 + row) * 128 + col] = acc[r];
    }
  }
}

// ---------------- Fusion + classifier (tiny; plain VALU) ----------------
__global__ __launch_bounds__(128, 1) void k_fusion(
    const float* __restrict__ g_mag, const float* __restrict__ g_phase,
    const float* p2m_w, const float* p2m_b, const float* m2p_w, const float* m2p_b,
    const float* out_w, const float* out_b, const float* cls_w, const float* cls_b,
    float* __restrict__ out)
{
  __shared__ float gm[128], gp[128], fin[256], meanv[128];
  const int b = blockIdx.x, d = threadIdx.x;
  float acc = 0.f;
  for (int t = 0; t < 128; ++t) {
    size_t row = (size_t)b * 128 + t;
    gm[d] = g_mag[row * 128 + d];
    gp[d] = g_phase[row * 128 + d];
    __syncthreads();
    float s1 = p2m_b[d], s2 = m2p_b[d];
    for (int k = 0; k < 128; ++k) {
      s1 += gp[k] * p2m_w[k * 128 + d];
      s2 += gm[k] * m2p_w[k * 128 + d];
    }
    fin[d]       = gm[d] * sigmoidf_(s1);
    fin[128 + d] = gp[d] * sigmoidf_(s2);
    __syncthreads();
    float f = out_b[d];
    for (int k = 0; k < 256; ++k) f += fin[k] * out_w[k * 128 + d];
    acc += f;
    __syncthreads();
  }
  meanv[d] = acc * (1.f / 128.f);
  __syncthreads();
  if (d < 11) {
    float o = cls_b[d];
    for (int k = 0; k < 128; ++k) o += meanv[k] * cls_w[k * 11 + d];
    out[b * 11 + d] = o;
  }
}

// ---------------- host ----------------
extern "C" void kernel_launch(void* const* d_in, const int* in_sizes, int n_in,
                              void* d_out, int out_size, void* d_ws, size_t ws_size,
                              hipStream_t stream) {
  (void)in_sizes; (void)n_in; (void)out_size; (void)ws_size;
  const float* zr     = (const float*)d_in[0];
  const float* zi     = (const float*)d_in[1];
  const int*   coords = (const int*)d_in[2];
  auto F = [&](int i) { return (const float*)d_in[i]; };
  // params flattened: sorted-key DFS of the params dict (JAX pytree order)
  MambaP gmP{F(19), F(20), F(21), F(22), F(23), F(24), F(25), F(26)};
  MambaP gpP{F(27), F(28), F(29), F(30), F(31), F(32), F(33), F(34)};
  MambaP lmP{F(37), F(38), F(39), F(40), F(41), F(42), F(43), F(44)};
  MambaP lpP{F(47), F(48), F(49), F(50), F(51), F(52), F(53), F(54)};

  char* ws = (char*)d_ws;
  const size_t TOKENS = 512ull * 256ull;            // 131072
  ushort_t* f_mag   = (ushort_t*)ws;  ws += TOKENS * 128 * sizeof(ushort_t);
  ushort_t* f_phase = (ushort_t*)ws;  ws += TOKENS * 128 * sizeof(ushort_t);
  uint_t*   idx0    = (uint_t*)ws;    ws += TOKENS * sizeof(uint_t);
  uint_t*   idx1    = (uint_t*)ws;    ws += TOKENS * sizeof(uint_t);
  float*    accum   = (float*)ws;     ws += TOKENS * 128 * sizeof(float);
  float*    seg_mag = (float*)ws;     ws += 512ull * 128 * sizeof(float);
  float*    seg_ph  = (float*)ws;     ws += 512ull * 128 * sizeof(float);
  float*    gmag    = (float*)ws;     ws += 512ull * 128 * sizeof(float);
  float*    gph     = (float*)ws;     ws += 512ull * 128 * sizeof(float);

  hipFuncSetAttribute(reinterpret_cast<const void*>(k_adapter),
                      hipFuncAttributeMaxDynamicSharedMemorySize, AD_SMEM);
  hipFuncSetAttribute(reinterpret_cast<const void*>(k_local_mamba),
                      hipFuncAttributeMaxDynamicSharedMemorySize, LM_SMEM);
  hipFuncSetAttribute(reinterpret_cast<const void*>(k_global_mamba),
                      hipFuncAttributeMaxDynamicSharedMemorySize, GM_SMEM);

  k_adapter<<<4096, 256, AD_SMEM, stream>>>(zr, zi, coords,
      F(4), F(3), F(6), F(5), F(10), F(9), F(8), F(7), f_mag, f_phase);
  k_sort<<<512, 256, 0, stream>>>(coords, idx0, idx1);

  // magnitude stream
  k_zero<<<1024, 256, 0, stream>>>(accum, TOKENS * 128);
  k_local_mamba<<<dim3(512, 2), 256, LM_SMEM, stream>>>(f_mag, idx0, idx1, lmP, accum);
  k_seg_reduce<<<512, 128, 0, stream>>>(accum, F(36), F(35), seg_mag);
  // phase stream
  k_zero<<<1024, 256, 0, stream>>>(accum, TOKENS * 128);
  k_local_mamba<<<dim3(512, 2), 256, LM_SMEM, stream>>>(f_phase, idx0, idx1, lpP, accum);
  k_seg_reduce<<<512, 128, 0, stream>>>(accum, F(46), F(45), seg_ph);

  k_global_mamba<<<dim3(4, 2), 256, GM_SMEM, stream>>>(seg_mag, seg_ph, gmP, gpP, gmag, gph);
  k_fusion<<<4, 128, 0, stream>>>(gmag, gph,
      F(18), F(17), F(14), F(13), F(16), F(15), F(12), F(11), (float*)d_out);
}
